// RobertaSelfAttention_21526376088228
// MI455X (gfx1250) — matrix operand order repaired and checked by reading, not compiled, not run on hardware
//
#include <hip/hip_runtime.h>

typedef __attribute__((ext_vector_type(16))) __bf16 v16bf;
typedef __attribute__((ext_vector_type(8)))  __bf16 v8bf;
typedef __attribute__((ext_vector_type(8)))  float  v8f;

#define BS   4
#define SEQ  2048
#define HIDN 1024
#define NHD  16
#define HD   64
#define MROW (BS * SEQ) /* 8192 */

#define CAT16(lo, hi) __builtin_shufflevector((lo), (hi), 0,1,2,3,4,5,6,7,8,9,10,11,12,13,14,15)

__device__ __forceinline__ __bf16 f2bf(float f) {
  unsigned u = __builtin_bit_cast(unsigned, f);
  unsigned r = u + 0x7FFFu + ((u >> 16) & 1u);   // round-to-nearest-even
  unsigned short h = (unsigned short)(r >> 16);
  return __builtin_bit_cast(__bf16, h);
}

__device__ __forceinline__ v8bf ld8(const __bf16* p) { return *(const v8bf*)p; }

#define WMMA_BF16(a, b, c) \
  __builtin_amdgcn_wmma_f32_16x16x32_bf16(false, (a), false, (b), (short)0, (c), false, false)

// ---------------------------------------------------------------------------
// f32 -> bf16 conversion pass (8 elements / thread, b128 in / b128 out)
// ---------------------------------------------------------------------------
__global__ void cvt_f32_to_bf16_v8(const float* __restrict__ s, __bf16* __restrict__ d, int n8) {
  int i = blockIdx.x * blockDim.x + threadIdx.x;
  if (i < n8) {
    const float4* sp = (const float4*)s + (size_t)i * 2;
    float4 f0 = sp[0], f1 = sp[1];
    v8bf o;
    o[0] = f2bf(f0.x); o[1] = f2bf(f0.y); o[2] = f2bf(f0.z); o[3] = f2bf(f0.w);
    o[4] = f2bf(f1.x); o[5] = f2bf(f1.y); o[6] = f2bf(f1.z); o[7] = f2bf(f1.w);
    ((v8bf*)d)[i] = o;
  }
}

// ---------------------------------------------------------------------------
// QKV projection: out = X @ W^T + bias, X:[M,K] bf16, W:[N,K] bf16.
// One wave computes a 32x64 output strip (8 accumulators). K loop unrolled by
// 2 with ping-pong operand register sets so loads for step k+1 are in flight
// while WMMAs for step k execute (no s_wait_loadcnt 0 stalls).
// transposed==0 -> store [bh, s, d]; transposed==1 -> store [bh, d, s] (for V)
// ---------------------------------------------------------------------------
__global__ __launch_bounds__(256)
void qkv_gemm_bf16(const __bf16* __restrict__ X, const __bf16* __restrict__ W,
                   const float* __restrict__ bias, __bf16* __restrict__ dst,
                   int transposed) {
  const int wid  = (blockIdx.x * blockDim.x + threadIdx.x) >> 5;
  const int lane = threadIdx.x & 31;
  const int ln = lane & 15, lh = lane >> 4;
  const int mt = wid >> 4;            // 256 m-tiles of 32 rows
  const int ng = wid & 15;            // 16 col groups of 64
  const int m0 = mt << 5, n0 = ng << 6;

  const __bf16* x0 = X + (size_t)(m0 + ln) * HIDN;       // rows m0 .. m0+15
  const __bf16* x1 = X + (size_t)(m0 + 16 + ln) * HIDN;  // rows m0+16 .. m0+31
  const __bf16* w0 = W + (size_t)(n0 + ln) * HIDN;

  auto ldA = [&](const __bf16* base, int k0) -> v16bf {
    int o = k0 + lh * 8;              // A layout: chunks {0-7,16-23}/{8-15,24-31}
    return CAT16(ld8(base + o), ld8(base + o + 16));
  };
  auto ldB = [&](int t, int k0) -> v16bf {
    int o = k0 + lh * 8;              // B layout: chunks {0-7,16-23}/{8-15,24-31}
    const __bf16* p = w0 + t * 16 * HIDN + o;
    return CAT16(ld8(p), ld8(p + 16));
  };

  v8f c00 = {}, c01 = {}, c02 = {}, c03 = {};
  v8f c10 = {}, c11 = {}, c12 = {}, c13 = {};

  // preload set A (k = 0)
  v16bf aA0 = ldA(x0, 0), aA1 = ldA(x1, 0);
  v16bf bA0 = ldB(0, 0), bA1 = ldB(1, 0), bA2 = ldB(2, 0), bA3 = ldB(3, 0);

#pragma unroll 1
  for (int k0 = 0; k0 < HIDN; k0 += 64) {
    // issue set B loads (k0+32) before consuming set A
    const int kB = k0 + 32;
    v16bf aB0 = ldA(x0, kB), aB1 = ldA(x1, kB);
    v16bf bB0 = ldB(0, kB), bB1 = ldB(1, kB), bB2 = ldB(2, kB), bB3 = ldB(3, kB);

    c00 = WMMA_BF16(aA0, bA0, c00);
    c01 = WMMA_BF16(aA0, bA1, c01);
    c02 = WMMA_BF16(aA0, bA2, c02);
    c03 = WMMA_BF16(aA0, bA3, c03);
    c10 = WMMA_BF16(aA1, bA0, c10);
    c11 = WMMA_BF16(aA1, bA1, c11);
    c12 = WMMA_BF16(aA1, bA2, c12);
    c13 = WMMA_BF16(aA1, bA3, c13);

    // issue set A loads for next iteration (wraps to 0 on last: harmless)
    const int kN = (k0 + 64) & (HIDN - 1);
    aA0 = ldA(x0, kN); aA1 = ldA(x1, kN);
    bA0 = ldB(0, kN); bA1 = ldB(1, kN); bA2 = ldB(2, kN); bA3 = ldB(3, kN);

    c00 = WMMA_BF16(aB0, bB0, c00);
    c01 = WMMA_BF16(aB0, bB1, c01);
    c02 = WMMA_BF16(aB0, bB2, c02);
    c03 = WMMA_BF16(aB0, bB3, c03);
    c10 = WMMA_BF16(aB1, bB0, c10);
    c11 = WMMA_BF16(aB1, bB1, c11);
    c12 = WMMA_BF16(aB1, bB2, c12);
    c13 = WMMA_BF16(aB1, bB3, c13);
  }

  const float bv0 = bias[n0 + ln];
  const float bv1 = bias[n0 + 16 + ln];
  const float bv2 = bias[n0 + 32 + ln];
  const float bv3 = bias[n0 + 48 + ln];

  auto store1 = [&](float v, int n, int bb, int s) {
    int h = n >> 6, d = n & 63;
    size_t bh = (size_t)bb * NHD + h;
    if (!transposed) dst[(bh * SEQ + s) * HD + d] = f2bf(v);
    else             dst[(bh * HD + d) * SEQ + s] = f2bf(v);
  };

#pragma unroll
  for (int r = 0; r < 8; ++r) {
    // C layout: VGPR r holds row r (lanes 0-15) / row r+8 (lanes 16-31)
    int m  = m0 + r + (lh << 3);
    int bb = m >> 11, s = m & (SEQ - 1);
    store1(c00[r] + bv0, n0 + ln,      bb, s);
    store1(c01[r] + bv1, n0 + 16 + ln, bb, s);
    store1(c02[r] + bv2, n0 + 32 + ln, bb, s);
    store1(c03[r] + bv3, n0 + 48 + ln, bb, s);
    int m2  = m + 16;
    int bb2 = m2 >> 11, s2 = m2 & (SEQ - 1);
    store1(c10[r] + bv0, n0 + ln,      bb2, s2);
    store1(c11[r] + bv1, n0 + 16 + ln, bb2, s2);
    store1(c12[r] + bv2, n0 + 32 + ln, bb2, s2);
    store1(c13[r] + bv3, n0 + 48 + ln, bb2, s2);
  }
}

// ---------------------------------------------------------------------------
// Flash attention: one wave per (b, h, 16-query tile), 32 keys / iteration.
// K tiles double-buffered (next iteration's K loads issued before softmax);
// V tiles loaded at iteration start so softmax + LDS round-trip hide them.
// ---------------------------------------------------------------------------
__global__ __launch_bounds__(256)
void flash_attn_bf16(const __bf16* __restrict__ Q, const __bf16* __restrict__ K,
                     const __bf16* __restrict__ Vt, const float* __restrict__ mask,
                     float* __restrict__ out) {
  __shared__ __attribute__((aligned(16))) __bf16 pt[8][16 * 32];

  const int wid  = (blockIdx.x * blockDim.x + threadIdx.x) >> 5;
  const int wib  = threadIdx.x >> 5;
  const int lane = threadIdx.x & 31;
  const int ln = lane & 15, lh = lane >> 4;
  const int qt = wid & (SEQ / 16 - 1);   // 128 query tiles
  const int bh = wid >> 7;               // 64 (b,h) pairs
  const int b  = bh >> 4;
  const int h  = bh & 15;

  const __bf16* Qh = Q  + (size_t)bh * SEQ * HD;
  const __bf16* Kh = K  + (size_t)bh * SEQ * HD;
  const __bf16* Vh = Vt + (size_t)bh * HD * SEQ;
  const float*  mk = mask + (size_t)b * SEQ;

  const int aoff = lh * 8;    // A-layout k-chunk base
  const int boff = lh * 8;    // B-layout k-chunk base

  const int qrow = (qt * 16 + ln) * HD;
  v16bf qa0 = CAT16(ld8(Qh + qrow + aoff),      ld8(Qh + qrow + 16 + aoff));
  v16bf qa1 = CAT16(ld8(Qh + qrow + 32 + aoff), ld8(Qh + qrow + 48 + aoff));

  v8f o0 = {}, o1 = {}, o2 = {}, o3 = {};
  float mrow[8], lrow[8];
#pragma unroll
  for (int r = 0; r < 8; ++r) { mrow[r] = -1e30f; lrow[r] = 0.f; }

  __bf16* myp = &pt[wib][0];

  auto ldK = [&](int j, int half, int c) -> v16bf {   // keys j+16*half.., d chunk c
    const __bf16* p = Kh + (size_t)(j + 16 * half + ln) * HD + 32 * c + boff;
    return CAT16(ld8(p), ld8(p + 16));
  };
  auto ldV = [&](int j, int t) -> v16bf {             // Vt[d = 16t+ln][j .. j+31]
    const __bf16* p = Vh + (size_t)(16 * t + ln) * SEQ + j + boff;
    return CAT16(ld8(p), ld8(p + 16));
  };

  // One 32-key step: consumes K set (kc*), writes next K set (kn*) for j+32.
  auto body = [&](int j,
                  v16bf& kc00, v16bf& kc01, v16bf& kc10, v16bf& kc11,
                  v16bf& kn00, v16bf& kn01, v16bf& kn10, v16bf& kn11) {
    // V tiles for this step (consumed only after softmax -> latency hidden)
    v16bf vb0 = ldV(j, 0), vb1 = ldV(j, 1), vb2 = ldV(j, 2), vb3 = ldV(j, 3);
    // K tiles for the next step (wraps on last iteration: harmless)
    const int jn = (j + 32) & (SEQ - 1);
    kn00 = ldK(jn, 0, 0); kn01 = ldK(jn, 0, 1);
    kn10 = ldK(jn, 1, 0); kn11 = ldK(jn, 1, 1);
    // stream-ahead prefetch (global_prefetch_b8)
    __builtin_prefetch(Kh + (size_t)((j + 128) & (SEQ - 1)) * HD, 0, 1);
    __builtin_prefetch(Vh + (size_t)ln * SEQ + ((j + 128) & (SEQ - 1)), 0, 1);

    // ---- S = Q * K^T over d=0..63 for keys [j, j+32) ----
    v8f s0 = {}, s1 = {};
    s0 = WMMA_BF16(qa0, kc00, s0);
    s0 = WMMA_BF16(qa1, kc01, s0);
    s1 = WMMA_BF16(qa0, kc10, s1);
    s1 = WMMA_BF16(qa1, kc11, s1);

    const float mv0 = mk[j + ln];
    const float mv1 = mk[j + 16 + ln];

    // ---- online softmax (rows live in 16-lane halves of the wave) ----
#pragma unroll
    for (int r = 0; r < 8; ++r) {
      float a0 = s0[r] * 0.125f + mv0;   // 1/sqrt(64)
      float a1 = s1[r] * 0.125f + mv1;
      float t = fmaxf(a0, a1);
      t = fmaxf(t, __shfl_xor(t, 1, 16));
      t = fmaxf(t, __shfl_xor(t, 2, 16));
      t = fmaxf(t, __shfl_xor(t, 4, 16));
      t = fmaxf(t, __shfl_xor(t, 8, 16));
      float mn    = fmaxf(mrow[r], t);
      float alpha = __expf(mrow[r] - mn);
      float p0 = __expf(a0 - mn);
      float p1 = __expf(a1 - mn);
      float rs = p0 + p1;
      rs += __shfl_xor(rs, 1, 16);
      rs += __shfl_xor(rs, 2, 16);
      rs += __shfl_xor(rs, 4, 16);
      rs += __shfl_xor(rs, 8, 16);
      lrow[r] = lrow[r] * alpha + rs;
      mrow[r] = mn;
      o0[r] *= alpha; o1[r] *= alpha; o2[r] *= alpha; o3[r] *= alpha;
      const int mr = r + (lh << 3);
      myp[mr * 32 + ln]      = f2bf(p0);   // C-layout -> LDS tile [m][n]
      myp[mr * 32 + 16 + ln] = f2bf(p1);
    }

    __builtin_amdgcn_wave_barrier();   // keep LDS store->load order (HW in-order per wave)
    v16bf pa = CAT16(ld8(myp + ln * 32 + aoff), ld8(myp + ln * 32 + 16 + aoff));
    __builtin_amdgcn_wave_barrier();

    // ---- O += P * V^T ----
    o0 = WMMA_BF16(pa, vb0, o0);
    o1 = WMMA_BF16(pa, vb1, o1);
    o2 = WMMA_BF16(pa, vb2, o2);
    o3 = WMMA_BF16(pa, vb3, o3);
  };

  // preload K set A for j = 0
  v16bf kA00 = ldK(0, 0, 0), kA01 = ldK(0, 0, 1);
  v16bf kA10 = ldK(0, 1, 0), kA11 = ldK(0, 1, 1);
  v16bf kB00, kB01, kB10, kB11;

#pragma unroll 1
  for (int j = 0; j < SEQ; j += 64) {
    body(j,      kA00, kA01, kA10, kA11, kB00, kB01, kB10, kB11);
    body(j + 32, kB00, kB01, kB10, kB11, kA00, kA01, kA10, kA11);
  }

  // ---- normalize and store context [B, S, HID] ----
#pragma unroll
  for (int r = 0; r < 8; ++r) {
    float inv = 1.0f / lrow[r];
    int s = qt * 16 + r + (lh << 3);
    float* op = out + ((size_t)(b * SEQ + s)) * HIDN + h * HD + ln;
    op[0]  = o0[r] * inv;
    op[16] = o1[r] * inv;
    op[32] = o2[r] * inv;
    op[48] = o3[r] * inv;
  }
}

// ---------------------------------------------------------------------------
extern "C" void kernel_launch(void* const* d_in, const int* in_sizes, int n_in,
                              void* d_out, int out_size, void* d_ws, size_t ws_size,
                              hipStream_t stream) {
  (void)in_sizes; (void)n_in; (void)out_size; (void)ws_size;
  const float* hidden = (const float*)d_in[0];
  const float* mask   = (const float*)d_in[1];
  const float* Wq = (const float*)d_in[2];
  const float* bq = (const float*)d_in[3];
  const float* Wk = (const float*)d_in[4];
  const float* bk = (const float*)d_in[5];
  const float* Wv = (const float*)d_in[6];
  const float* bv = (const float*)d_in[7];
  float* out = (float*)d_out;

  char* p = (char*)d_ws;
  __bf16* Xb  = (__bf16*)p; p += (size_t)MROW * HIDN * 2;  // 16 MB
  __bf16* Wqb = (__bf16*)p; p += (size_t)HIDN * HIDN * 2;  //  2 MB
  __bf16* Wkb = (__bf16*)p; p += (size_t)HIDN * HIDN * 2;
  __bf16* Wvb = (__bf16*)p; p += (size_t)HIDN * HIDN * 2;
  __bf16* Qb  = (__bf16*)p; p += (size_t)MROW * HIDN * 2;  // [bh, s, d]
  __bf16* Kb  = (__bf16*)p; p += (size_t)MROW * HIDN * 2;  // [bh, s, d]
  __bf16* Vtb = (__bf16*)p;                                // [bh, d, s]

  const int nh8 = (MROW * HIDN) / 8;
  const int nw8 = (HIDN * HIDN) / 8;
  cvt_f32_to_bf16_v8<<<(nh8 + 255) / 256, 256, 0, stream>>>(hidden, Xb, nh8);
  cvt_f32_to_bf16_v8<<<(nw8 + 255) / 256, 256, 0, stream>>>(Wq, Wqb, nw8);
  cvt_f32_to_bf16_v8<<<(nw8 + 255) / 256, 256, 0, stream>>>(Wk, Wkb, nw8);
  cvt_f32_to_bf16_v8<<<(nw8 + 255) / 256, 256, 0, stream>>>(Wv, Wvb, nw8);

  // 4096 waves per GEMM (256 row tiles x 16 col groups), 8 waves / block
  qkv_gemm_bf16<<<512, 256, 0, stream>>>(Xb, Wqb, bq, Qb, 0);
  qkv_gemm_bf16<<<512, 256, 0, stream>>>(Xb, Wkb, bk, Kb, 0);
  qkv_gemm_bf16<<<512, 256, 0, stream>>>(Xb, Wvb, bv, Vtb, 1);

  // 8192 waves (64 bh x 128 query tiles), 8 waves / block
  flash_attn_bf16<<<1024, 256, 0, stream>>>(Qb, Kb, Vtb, mask, out);
}
